// GOALRNN_31396210934229
// MI455X (gfx1250) — compile-verified, matrix operand order checked by reading, not run on hardware
//
#include <hip/hip_runtime.h>
#include <hip/hip_bf16.h>
#include <cstdint>

// ---------------------------------------------------------------------------
// GOALRNN for MI455X (gfx1250, wave32, WMMA).
//
// Roofline: ~151 GFLOP total, ~70MB inputs -> latency bound on the T=1024
// sequential LSTM scan, not HBM bound.  Strategy:
//   K1 prep   : f32->f16 weight conversion (+K padding to multiples of 32)
//   K2 lstm   : ONE persistent workgroup (1024 thr = 32 waves on one WGP).
//               h-state double-buffered in LDS as f16 (4 x 64KB = 256KB of
//               the 320KB WGP LDS); c-state + biases live in VGPRs across
//               all 1024 steps.  Each wave's 4 tasks share one M-tile and
//               4 consecutive H-tiles, so each weight matrix needs exactly
//               ONE per-lane base pointer: task/gate/k-tile selection is
//               folded into the 24-bit instruction offsets (no spilled
//               address pairs).  GEMMs via v_wmma_f32_16x16x32_f16; per
//               k-tile the 4 gate B-fragments are batch-loaded so loads
//               clause and 4 WMMAs issue back-to-back.
//   K3 heads  : embarrassingly parallel over T*B rows; one 16-row tile per
//               wave, 117 WMMAs/tile, scalar tail for tanh/sampling/logprob.
// ---------------------------------------------------------------------------

#define T_STEPS 1024
#define BB      256
#define HH      128      // hidden per layer
#define G4      512      // 4*H gate width
#define OBS_D   64

typedef __attribute__((ext_vector_type(16))) _Float16 v16h;
typedef __attribute__((ext_vector_type(8)))  _Float16 v8h;
typedef __attribute__((ext_vector_type(8)))  float    v8f;

// ---- workspace layout (bytes) ---------------------------------------------
static constexpr size_t LAT_OFF  = 0;                                   // f16 latents (T*B x 128)
static constexpr size_t WI0_OFF  = (size_t)T_STEPS * BB * HH * 2;       // 67,108,864
static constexpr size_t WH0_OFF  = WI0_OFF + 512 * 96 * 2;
static constexpr size_t WI1_OFF  = WH0_OFF + 512 * 128 * 2;
static constexpr size_t WH1_OFF  = WI1_OFF + 512 * 128 * 2;
static constexpr size_t GD1_OFF  = WH1_OFF + 512 * 128 * 2;
static constexpr size_t GD2_OFF  = GD1_OFF + 64 * 64 * 2;
static constexpr size_t AD1_OFF  = GD2_OFF + 32 * 64 * 2;
static constexpr size_t AD2_OFF  = AD1_OFF + 80 * 96 * 2;
static constexpr size_t VD1_OFF  = AD2_OFF + 16 * 96 * 2;
static constexpr size_t LPD1_OFF = VD1_OFF + 80 * 96 * 2;

// ---- output layout (floats, reference return order) -----------------------
static constexpr size_t NTB     = (size_t)T_STEPS * BB;     // 262144
static constexpr size_t A_OFF   = 0;                        // actions   (T,B,8)
static constexpr size_t LPA_OFF = A_OFF + NTB * 8;          // lp_actions(T,B)
static constexpr size_t G_OFF   = LPA_OFF + NTB;            // goals     (T,B,16)
static constexpr size_t LPG_OFF = G_OFF + NTB * 16;         // lp_goals  (T,B)
static constexpr size_t V_OFF   = LPG_OFF + NTB;            // values    (T,B,1)
static constexpr size_t LPV_OFF = V_OFF + NTB;              // lp_values (T,B,1)

#define LOG2PI 1.8378770664093453f

// ---------------------------------------------------------------------------
// helpers
// ---------------------------------------------------------------------------
__device__ __forceinline__ v8f vzero8() {
  v8f z;
#pragma unroll
  for (int i = 0; i < 8; ++i) z[i] = 0.0f;
  return z;
}

__device__ __forceinline__ float sigm(float x) { return 1.0f / (1.0f + __expf(-x)); }

// intra-wave LDS producer->consumer fence (lanes are lockstep; only need the
// DS counter drained).  CDNA5 split counters: s_wait_dscnt.
__device__ __forceinline__ void lds_fence() {
  asm volatile("s_wait_dscnt 0" ::: "memory");
}

// Fragment load from a precomputed per-lane base + CONSTANT element offset
// (offset folds into the instruction's immediate).  Layout per ISA 7.12.2:
// the 16 halves per lane are two contiguous runs of 8 (chunks at +0, +16).
__device__ __forceinline__ v16h frag_at(const _Float16* q, int eoff) {
  v8h c0 = *(const v8h*)(q + eoff);
  v8h c1 = *(const v8h*)(q + eoff + 16);
  v16h o;
#pragma unroll
  for (int i = 0; i < 8; ++i) { o[i] = c0[i]; o[i + 8] = c1[i]; }
  return o;
}

// Same but sourced from f32 data (converted on load).
__device__ __forceinline__ v16h frag32_at(const float* __restrict__ q, int eoff) {
  float4 a0 = *(const float4*)(q + eoff);
  float4 a1 = *(const float4*)(q + eoff + 4);
  float4 a2 = *(const float4*)(q + eoff + 16);
  float4 a3 = *(const float4*)(q + eoff + 20);
  v16h o;
  o[0]  = (_Float16)a0.x; o[1]  = (_Float16)a0.y; o[2]  = (_Float16)a0.z; o[3]  = (_Float16)a0.w;
  o[4]  = (_Float16)a1.x; o[5]  = (_Float16)a1.y; o[6]  = (_Float16)a1.z; o[7]  = (_Float16)a1.w;
  o[8]  = (_Float16)a2.x; o[9]  = (_Float16)a2.y; o[10] = (_Float16)a2.z; o[11] = (_Float16)a2.w;
  o[12] = (_Float16)a3.x; o[13] = (_Float16)a3.y; o[14] = (_Float16)a3.z; o[15] = (_Float16)a3.w;
  return o;
}

// General row-major fragment load (used by the heads kernel).
__device__ __forceinline__ v16h frag_ld_f16(const _Float16* p, int ld, int row0, int k0) {
  const int l = threadIdx.x & 31;
  const _Float16* q = p + (size_t)(row0 + (l & 15)) * ld + k0 + 8 * (l >> 4);
  return frag_at(q, 0);
}

// Rank-1 K-tile for the lone `lps` input column (K-local 0 only).
__device__ __forceinline__ v16h frag_ld_lps(const float* __restrict__ lps_t, int row0) {
  const int l = threadIdx.x & 31;
  v16h o;
#pragma unroll
  for (int i = 0; i < 16; ++i) o[i] = (_Float16)0.0f;
  if (l < 16) o[0] = (_Float16)lps_t[row0 + l];
  return o;
}

__device__ __forceinline__ v8f wmma16(v16h a, v16h b, v8f c) {
  // D = A(16x32) x B(32x16) + C ; emits v_wmma_f32_16x16x32_f16
  return __builtin_amdgcn_wmma_f32_16x16x32_f16(false, a, false, b, (short)0, c, false, false);
}

// ---------------------------------------------------------------------------
// K1: weight conversion to f16 (K padded to multiples of 32 where needed)
// ---------------------------------------------------------------------------
__global__ void prep_weights_kernel(
    const float* __restrict__ w_ih0, const float* __restrict__ w_hh0,
    const float* __restrict__ w_ih1, const float* __restrict__ w_hh1,
    const float* __restrict__ gd_w1, const float* __restrict__ gd_w2,
    const float* __restrict__ ad_w1, const float* __restrict__ ad_w2,
    const float* __restrict__ vd_w1, const float* __restrict__ lpd_w1,
    char* __restrict__ ws)
{
  const int tid = blockIdx.x * blockDim.x + threadIdx.x;
  const int nth = gridDim.x * blockDim.x;
  _Float16* wi0 = (_Float16*)(ws + WI0_OFF);
  _Float16* wh0 = (_Float16*)(ws + WH0_OFF);
  _Float16* wi1 = (_Float16*)(ws + WI1_OFF);
  _Float16* wh1 = (_Float16*)(ws + WH1_OFF);
  _Float16* gw1 = (_Float16*)(ws + GD1_OFF);
  _Float16* gw2 = (_Float16*)(ws + GD2_OFF);
  _Float16* aw1 = (_Float16*)(ws + AD1_OFF);
  _Float16* aw2 = (_Float16*)(ws + AD2_OFF);
  _Float16* vw1 = (_Float16*)(ws + VD1_OFF);
  _Float16* lw1 = (_Float16*)(ws + LPD1_OFF);

  for (int i = tid; i < 512 * 96; i += nth) { int n = i / 96, k = i - n * 96; wi0[i] = (_Float16)(k < 65 ? w_ih0[n * 65 + k] : 0.0f); }
  for (int i = tid; i < 512 * 128; i += nth) wh0[i] = (_Float16)w_hh0[i];
  for (int i = tid; i < 512 * 128; i += nth) wi1[i] = (_Float16)w_ih1[i];
  for (int i = tid; i < 512 * 128; i += nth) wh1[i] = (_Float16)w_hh1[i];
  for (int i = tid; i < 64 * 64;  i += nth) gw1[i] = (_Float16)gd_w1[i];
  for (int i = tid; i < 32 * 64;  i += nth) gw2[i] = (_Float16)gd_w2[i];
  for (int i = tid; i < 80 * 96;  i += nth) { int n = i / 96, k = i - n * 96; aw1[i] = (_Float16)(k < 80 ? ad_w1[n * 80 + k] : 0.0f); }
  for (int i = tid; i < 16 * 96;  i += nth) { int n = i / 96, k = i - n * 96; aw2[i] = (_Float16)(k < 80 ? ad_w2[n * 80 + k] : 0.0f); }
  for (int i = tid; i < 80 * 96;  i += nth) { int n = i / 96, k = i - n * 96; vw1[i] = (_Float16)(k < 80 ? vd_w1[n * 80 + k] : 0.0f); }
  for (int i = tid; i < 192 * 192; i += nth) lw1[i] = (_Float16)lpd_w1[i];
}

// ---------------------------------------------------------------------------
// K2: sequential 2-layer LSTM scan.  One workgroup, 32 waves.
// 128 tasks (16 M-tiles x 8 H-tiles); wave wv owns M-tile (wv>>1) and
// H-tiles ht0..ht0+3 with ht0 = 4*(wv&1).  A task = one 16x16 tile of each
// of the 4 gates.  All weight addressing goes through ONE per-lane base
// pointer per matrix with constant folded offsets.
// ---------------------------------------------------------------------------
__global__ void __launch_bounds__(1024)
lstm_seq_kernel(const float* __restrict__ obs, const float* __restrict__ lps,
                const float* __restrict__ h0, const float* __restrict__ c0,
                const float* __restrict__ b_ih0, const float* __restrict__ b_hh0,
                const float* __restrict__ b_ih1, const float* __restrict__ b_hh1,
                char* __restrict__ ws)
{
  extern __shared__ char smem[];
  _Float16* bufA0 = (_Float16*)smem;               // layer0 h ping : 256x128 f16
  _Float16* bufA1 = (_Float16*)(smem + 65536);     // layer0 h pong
  _Float16* bufB0 = (_Float16*)(smem + 131072);    // layer1 h ping
  _Float16* bufB1 = (_Float16*)(smem + 196608);    // layer1 h pong

  const _Float16* wi0 = (const _Float16*)(ws + WI0_OFF);   // 512 x 96
  const _Float16* wh0 = (const _Float16*)(ws + WH0_OFF);   // 512 x 128
  const _Float16* wi1 = (const _Float16*)(ws + WI1_OFF);   // 512 x 128
  const _Float16* wh1 = (const _Float16*)(ws + WH1_OFF);   // 512 x 128
  _Float16* lat = (_Float16*)(ws + LAT_OFF);               // (T*B) x 128

  const int tid    = threadIdx.x;
  const int wv     = tid >> 5;
  const int lane   = tid & 31;
  const int g      = lane >> 4;
  const int nn     = lane & 15;
  const int lane15 = lane & 15;

  const int mtw = wv >> 1;          // wave's M tile (same for its 4 tasks)
  const int ht0 = (wv & 1) * 4;     // first of 4 consecutive H tiles

  // initial hidden state into the "current" buffers
  for (int i = tid; i < BB * HH; i += 1024) {
    bufA0[i] = (_Float16)h0[i];
    bufB0[i] = (_Float16)h0[BB * HH + i];
  }

  // persistent cell state + hoisted combined biases (t-invariant)
  float c0r[4][8], c1r[4][8];
  float bs0[4][4], bs1[4][4];
#pragma unroll
  for (int i = 0; i < 4; ++i) {
    const int col = (ht0 + i) * 16 + nn;
#pragma unroll
    for (int gi = 0; gi < 4; ++gi) {
      bs0[i][gi] = b_ih0[gi * HH + col] + b_hh0[gi * HH + col];
      bs1[i][gi] = b_ih1[gi * HH + col] + b_hh1[gi * HH + col];
    }
#pragma unroll
    for (int r = 0; r < 8; ++r) {
      const int b = mtw * 16 + r + 8 * g;
      c0r[i][r] = c0[(size_t)b * HH + col];
      c1r[i][r] = c0[(size_t)BB * HH + (size_t)b * HH + col];
    }
  }

  // ONE loop-invariant per-lane base pointer per weight matrix.
  // Offsets at use sites: ld=96  -> gi*12288 + i*1536 + kt
  //                       ld=128 -> gi*16384 + i*2048 + kt   (all constants)
  const _Float16* qwi0 = wi0 + (ht0 * 16 + lane15) * 96  + 8 * g;
  const _Float16* qwh0 = wh0 + (ht0 * 16 + lane15) * 128 + 8 * g;
  const _Float16* qwi1 = wi1 + (ht0 * 16 + lane15) * 128 + 8 * g;
  const _Float16* qwh1 = wh1 + (ht0 * 16 + lane15) * 128 + 8 * g;

  _Float16* sAc = bufA0; _Float16* sAn = bufA1;   // layer0 cur / next
  _Float16* sBc = bufB0; _Float16* sBn = bufB1;   // layer1 cur / next
  __syncthreads();

  for (int t = 0; t < T_STEPS; ++t) {
    // per-step per-lane bases (few VALU ops, not worth keeping live)
    const float*    qobs = obs + (size_t)t * BB * OBS_D + (mtw * 16 + lane15) * OBS_D + 8 * g;
    const float*    lps_t = lps + (size_t)t * BB;
    const _Float16* qAc  = sAc + (mtw * 16 + lane15) * HH + 8 * g;
    const _Float16* qBc  = sBc + (mtw * 16 + lane15) * HH + 8 * g;
    const _Float16* qAn  = sAn + (mtw * 16 + lane15) * HH + 8 * g;
    _Float16*       pAn  = sAn + (mtw * 16 + 8 * g) * HH + ht0 * 16 + nn;
    _Float16*       pBn  = sBn + (mtw * 16 + 8 * g) * HH + ht0 * 16 + nn;
    _Float16*       plat = lat + ((size_t)t * BB + mtw * 16 + 8 * g) * HH + ht0 * 16 + nn;

    // prefetch next step's observations while this step computes
    if (t + 1 < T_STEPS)
      __builtin_prefetch(obs + (size_t)(t + 1) * BB * OBS_D + tid * 16, 0, 0);

    // ---------------- layer 0: g0 = x@Wih0^T + hA@Whh0^T + b --------------
#pragma unroll
    for (int i = 0; i < 4; ++i) {
      v8f acc[4];
#pragma unroll
      for (int gi = 0; gi < 4; ++gi) acc[gi] = vzero8();

      // obs part, K = 64 : batch the 4 gate B-fragments, then 4 WMMAs
#pragma unroll
      for (int kt = 0; kt < 64; kt += 32) {
        const v16h a = frag32_at(qobs, kt);
        v16h bf[4];
#pragma unroll
        for (int gi = 0; gi < 4; ++gi) bf[gi] = frag_at(qwi0, gi * 12288 + i * 1536 + kt);
#pragma unroll
        for (int gi = 0; gi < 4; ++gi) acc[gi] = wmma16(a, bf[gi], acc[gi]);
      }
      // lps column (rank-1 K-tile at k0 = 64; weights padded with zeros)
      {
        const v16h a = frag_ld_lps(lps_t, mtw * 16);
        v16h bf[4];
#pragma unroll
        for (int gi = 0; gi < 4; ++gi) bf[gi] = frag_at(qwi0, gi * 12288 + i * 1536 + 64);
#pragma unroll
        for (int gi = 0; gi < 4; ++gi) acc[gi] = wmma16(a, bf[gi], acc[gi]);
      }
      // recurrent part, K = 128
#pragma unroll
      for (int kt = 0; kt < 128; kt += 32) {
        const v16h a = frag_at(qAc, kt);
        v16h bf[4];
#pragma unroll
        for (int gi = 0; gi < 4; ++gi) bf[gi] = frag_at(qwh0, gi * 16384 + i * 2048 + kt);
#pragma unroll
        for (int gi = 0; gi < 4; ++gi) acc[gi] = wmma16(a, bf[gi], acc[gi]);
      }
      // activations -> store new h straight into the "next" buffer
#pragma unroll
      for (int r = 0; r < 8; ++r) {
        const float iv = sigm(acc[0][r] + bs0[i][0]);
        const float fv = sigm(acc[1][r] + bs0[i][1]);
        const float gv = tanhf(acc[2][r] + bs0[i][2]);
        const float ov = sigm(acc[3][r] + bs0[i][3]);
        const float cc = fv * c0r[i][r] + iv * gv;
        c0r[i][r] = cc;
        pAn[r * HH + i * 16] = (_Float16)(ov * tanhf(cc));
      }
    }
    __syncthreads();   // new layer-0 h complete

    // ---------------- layer 1: g1 = hA@Wih1^T + hB@Whh1^T + b -------------
#pragma unroll
    for (int i = 0; i < 4; ++i) {
      v8f acc[4];
#pragma unroll
      for (int gi = 0; gi < 4; ++gi) acc[gi] = vzero8();
#pragma unroll
      for (int kt = 0; kt < 128; kt += 32) {
        const v16h a = frag_at(qAn, kt);
        v16h bf[4];
#pragma unroll
        for (int gi = 0; gi < 4; ++gi) bf[gi] = frag_at(qwi1, gi * 16384 + i * 2048 + kt);
#pragma unroll
        for (int gi = 0; gi < 4; ++gi) acc[gi] = wmma16(a, bf[gi], acc[gi]);
      }
#pragma unroll
      for (int kt = 0; kt < 128; kt += 32) {
        const v16h a = frag_at(qBc, kt);
        v16h bf[4];
#pragma unroll
        for (int gi = 0; gi < 4; ++gi) bf[gi] = frag_at(qwh1, gi * 16384 + i * 2048 + kt);
#pragma unroll
        for (int gi = 0; gi < 4; ++gi) acc[gi] = wmma16(a, bf[gi], acc[gi]);
      }
#pragma unroll
      for (int r = 0; r < 8; ++r) {
        const float iv = sigm(acc[0][r] + bs1[i][0]);
        const float fv = sigm(acc[1][r] + bs1[i][1]);
        const float gv = tanhf(acc[2][r] + bs1[i][2]);
        const float ov = sigm(acc[3][r] + bs1[i][3]);
        const float cc = fv * c1r[i][r] + iv * gv;
        c1r[i][r] = cc;
        const _Float16 hv = (_Float16)(ov * tanhf(cc));
        pBn[r * HH + i * 16] = hv;
        plat[r * HH + i * 16] = hv;   // latents_full (f16)
      }
    }
    __syncthreads();   // new layer-1 h complete

    // ping-pong
    _Float16* tp;
    tp = sAc; sAc = sAn; sAn = tp;
    tp = sBc; sBc = sBn; sBn = tp;
  }
}

// ---------------------------------------------------------------------------
// K3: per-timestep heads.  blockDim = 64 (2 waves); each wave = one 16-row
// tile of the flattened (T*B) x ... problem.  Per-wave LDS scratch = 17408B.
// ---------------------------------------------------------------------------
__global__ void __launch_bounds__(64)
heads_kernel(const float* __restrict__ obs,
             const float* __restrict__ geps, const float* __restrict__ aeps,
             const float* __restrict__ gd_b1, const float* __restrict__ gd_b2,
             const float* __restrict__ ad_b1, const float* __restrict__ ad_b2,
             const float* __restrict__ vd_b1, const float* __restrict__ vd_w2,
             const float* __restrict__ vd_b2,
             const float* __restrict__ lpd_b1, const float* __restrict__ lpd_w2,
             const float* __restrict__ lpd_b2,
             const char* __restrict__ ws, float* __restrict__ out)
{
  extern __shared__ char smem[];
  const int wv   = threadIdx.x >> 5;
  const int lane = threadIdx.x & 31;
  const int g    = lane >> 4;
  const int nn   = lane & 15;

  char* base = smem + (size_t)wv * 17408;
  _Float16* s_lo  = (_Float16*)base;            // 16 x 192 f16 : [latents(128)|obs(64)]
  _Float16* s_hid = (_Float16*)(base + 6144);   // 16 x 192 f16 : hidden scratch
  _Float16* s_x2  = (_Float16*)(base + 12288);  // 16 x 96  f16 : [goals(16)|obs(64)|pad]
  float*    s_go  = (float*)   (base + 15360);  // 16 x 32  f32 : small gemm outputs

  const _Float16* lat = (const _Float16*)(ws + LAT_OFF);
  const _Float16* gw1 = (const _Float16*)(ws + GD1_OFF);
  const _Float16* gw2 = (const _Float16*)(ws + GD2_OFF);
  const _Float16* aw1 = (const _Float16*)(ws + AD1_OFF);
  const _Float16* aw2 = (const _Float16*)(ws + AD2_OFF);
  const _Float16* vw1 = (const _Float16*)(ws + VD1_OFF);
  const _Float16* lw1 = (const _Float16*)(ws + LPD1_OFF);

  const int r0 = (blockIdx.x * 2 + wv) * 16;    // first global row of tile

  // ---- stage [latents | obs] ----------------------------------------------
  for (int idx = lane; idx < 16 * 192; idx += 32) {
    const int row = idx / 192, k = idx - row * 192;
    s_lo[idx] = (k < 128) ? lat[(size_t)(r0 + row) * HH + k]
                          : (_Float16)obs[(size_t)(r0 + row) * OBS_D + (k - 128)];
  }
  lds_fence();

  // ---- goal decoder: relu(lat64 @ W1^T + b1) @ W2^T + b2 ------------------
#pragma unroll
  for (int nt = 0; nt < 4; ++nt) {
    v8f acc = vzero8();
#pragma unroll
    for (int kt = 0; kt < 64; kt += 32)
      acc = wmma16(frag_ld_f16(s_lo, 192, 0, kt), frag_ld_f16(gw1, 64, nt * 16, kt), acc);
    const int col = nt * 16 + nn;
    const float bb = gd_b1[col];
#pragma unroll
    for (int r = 0; r < 8; ++r) {
      float hv = acc[r] + bb; hv = hv > 0.0f ? hv : 0.0f;
      s_hid[(r + 8 * g) * 192 + col] = (_Float16)hv;
    }
  }
  lds_fence();
#pragma unroll
  for (int nt = 0; nt < 2; ++nt) {
    v8f acc = vzero8();
#pragma unroll
    for (int kt = 0; kt < 64; kt += 32)
      acc = wmma16(frag_ld_f16(s_hid, 192, 0, kt), frag_ld_f16(gw2, 64, nt * 16, kt), acc);
    const int col = nt * 16 + nn;
    const float bb = gd_b2[col];
#pragma unroll
    for (int r = 0; r < 8; ++r) s_go[(r + 8 * g) * 32 + col] = acc[r] + bb;
  }
  // fill obs part + zero pad of x2 while gd output settles
  for (int idx = lane; idx < 16 * 96; idx += 32) {
    const int row = idx / 96, k = idx - row * 96;
    if (k >= 16) s_x2[idx] = (k < 80) ? s_lo[row * 192 + 128 + (k - 16)] : (_Float16)0.0f;
  }
  lds_fence();

  // ---- sample goals + log-prob (scalar tail, one row per lane) ------------
  if (lane < 16) {
    const int i = lane; const size_t r = (size_t)r0 + i;
    float s = 0.0f;
#pragma unroll
    for (int d = 0; d < 16; ++d) {
      const float gmr = s_go[i * 32 + d], gsr = s_go[i * 32 + 16 + d];
      const float gm  = 100.0f * tanhf(gmr * 0.001f);
      const float gs  = tanhf(gsr);
      const float gv  = gs * gs + 0.001f;
      const float gl  = gm + __fsqrt_rn(gv) * geps[r * 16 + d];
      out[G_OFF + r * 16 + d] = gl;
      const float dd = gl - gm;
      s += dd * dd / gv + __logf(gv);
      s_x2[i * 96 + d] = (_Float16)gl;
    }
    out[LPG_OFF + r] = -0.5f * (s + 16.0f * LOG2PI);
  }
  lds_fence();

  // ---- action decoder: relu(x2 @ W1^T + b1) @ W2^T + b2 -------------------
#pragma unroll
  for (int nt = 0; nt < 5; ++nt) {
    v8f acc = vzero8();
#pragma unroll
    for (int kt = 0; kt < 96; kt += 32)
      acc = wmma16(frag_ld_f16(s_x2, 96, 0, kt), frag_ld_f16(aw1, 96, nt * 16, kt), acc);
    const int col = nt * 16 + nn;
    const float bb = ad_b1[col];
#pragma unroll
    for (int r = 0; r < 8; ++r) {
      float hv = acc[r] + bb; hv = hv > 0.0f ? hv : 0.0f;
      s_hid[(r + 8 * g) * 192 + col] = (_Float16)hv;
    }
  }
  for (int idx = lane; idx < 16 * 16; idx += 32)         // zero pad K 80..95
    s_hid[(idx / 16) * 192 + 80 + (idx & 15)] = (_Float16)0.0f;
  lds_fence();
  {
    v8f acc = vzero8();
#pragma unroll
    for (int kt = 0; kt < 96; kt += 32)
      acc = wmma16(frag_ld_f16(s_hid, 192, 0, kt), frag_ld_f16(aw2, 96, 0, kt), acc);
    const float bb = ad_b2[nn];
#pragma unroll
    for (int r = 0; r < 8; ++r) s_go[(r + 8 * g) * 32 + nn] = acc[r] + bb;
  }
  lds_fence();
  if (lane < 16) {
    const int i = lane; const size_t r = (size_t)r0 + i;
    float s = 0.0f;
#pragma unroll
    for (int d = 0; d < 8; ++d) {
      const float am = s_go[i * 32 + d], as = s_go[i * 32 + 8 + d];
      const float av = as * as + 0.001f;
      const float ac = am + __fsqrt_rn(av) * aeps[r * 8 + d];
      out[A_OFF + r * 8 + d] = ac;
      const float dd = ac - am;
      s += dd * dd / av + __logf(av);
    }
    out[LPA_OFF + r] = -0.5f * (s + 8.0f * LOG2PI);
  }
  lds_fence();

  // ---- value decoder ------------------------------------------------------
#pragma unroll
  for (int nt = 0; nt < 5; ++nt) {
    v8f acc = vzero8();
#pragma unroll
    for (int kt = 0; kt < 96; kt += 32)
      acc = wmma16(frag_ld_f16(s_x2, 96, 0, kt), frag_ld_f16(vw1, 96, nt * 16, kt), acc);
    const int col = nt * 16 + nn;
    const float bb = vd_b1[col];
#pragma unroll
    for (int r = 0; r < 8; ++r) {
      float hv = acc[r] + bb; hv = hv > 0.0f ? hv : 0.0f;
      s_hid[(r + 8 * g) * 192 + col] = (_Float16)hv;
    }
  }
  lds_fence();
  if (lane < 16) {
    const int i = lane; const size_t r = (size_t)r0 + i;
    float v = vd_b2[0];
    for (int k = 0; k < 80; ++k) v += (float)s_hid[i * 192 + k] * vd_w2[k];
    out[V_OFF + r] = v;
  }
  lds_fence();

  // ---- lp decoder: relu([lat|obs] @ W1^T + b1) . w2 + b2 ------------------
#pragma unroll
  for (int nt = 0; nt < 12; ++nt) {
    v8f acc = vzero8();
#pragma unroll
    for (int kt = 0; kt < 192; kt += 32)
      acc = wmma16(frag_ld_f16(s_lo, 192, 0, kt), frag_ld_f16(lw1, 192, nt * 16, kt), acc);
    const int col = nt * 16 + nn;
    const float bb = lpd_b1[col];
#pragma unroll
    for (int r = 0; r < 8; ++r) {
      float hv = acc[r] + bb; hv = hv > 0.0f ? hv : 0.0f;
      s_hid[(r + 8 * g) * 192 + col] = (_Float16)hv;
    }
  }
  lds_fence();
  if (lane < 16) {
    const int i = lane; const size_t r = (size_t)r0 + i;
    float v = lpd_b2[0];
    for (int k = 0; k < 192; ++k) v += (float)s_hid[i * 192 + k] * lpd_w2[k];
    out[LPV_OFF + r] = v;
  }
}

// ---------------------------------------------------------------------------
extern "C" void kernel_launch(void* const* d_in, const int* in_sizes, int n_in,
                              void* d_out, int out_size, void* d_ws, size_t ws_size,
                              hipStream_t stream)
{
  const float* obs    = (const float*)d_in[0];
  const float* lps    = (const float*)d_in[1];
  const float* geps   = (const float*)d_in[2];
  const float* aeps   = (const float*)d_in[3];
  const float* h0     = (const float*)d_in[4];
  const float* c0     = (const float*)d_in[5];
  const float* w_ih0  = (const float*)d_in[6];
  const float* b_ih0  = (const float*)d_in[7];
  const float* w_hh0  = (const float*)d_in[8];
  const float* b_hh0  = (const float*)d_in[9];
  const float* w_ih1  = (const float*)d_in[10];
  const float* b_ih1  = (const float*)d_in[11];
  const float* w_hh1  = (const float*)d_in[12];
  const float* b_hh1  = (const float*)d_in[13];
  const float* gd_w1  = (const float*)d_in[14];
  const float* gd_b1  = (const float*)d_in[15];
  const float* gd_w2  = (const float*)d_in[16];
  const float* gd_b2  = (const float*)d_in[17];
  const float* ad_w1  = (const float*)d_in[18];
  const float* ad_b1  = (const float*)d_in[19];
  const float* ad_w2  = (const float*)d_in[20];
  const float* ad_b2  = (const float*)d_in[21];
  const float* vd_w1  = (const float*)d_in[22];
  const float* vd_b1  = (const float*)d_in[23];
  const float* vd_w2  = (const float*)d_in[24];
  const float* vd_b2  = (const float*)d_in[25];
  const float* lpd_w1 = (const float*)d_in[26];
  const float* lpd_b1 = (const float*)d_in[27];
  const float* lpd_w2 = (const float*)d_in[28];
  const float* lpd_b2 = (const float*)d_in[29];

  char*  ws  = (char*)d_ws;
  float* out = (float*)d_out;

  prep_weights_kernel<<<128, 256, 0, stream>>>(
      w_ih0, w_hh0, w_ih1, w_hh1, gd_w1, gd_w2, ad_w1, ad_w2, vd_w1, lpd_w1, ws);

  // one persistent workgroup, 32 waves, 256KB dynamic LDS (of 320KB WGP LDS)
  lstm_seq_kernel<<<1, 1024, 262144, stream>>>(
      obs, lps, h0, c0, b_ih0, b_hh0, b_ih1, b_hh1, ws);

  // 16384 row tiles, 2 waves/block, 34816B dynamic LDS per block
  heads_kernel<<<8192, 64, 34816, stream>>>(
      obs, geps, aeps, gd_b1, gd_b2, ad_b1, ad_b2,
      vd_b1, vd_w2, vd_b2, lpd_b1, lpd_w2, lpd_b2, ws, out);
}
